// SpikingSSMMoEBlock_49374944034996
// MI455X (gfx1250) — matrix operand order repaired and checked by reading, not compiled, not run on hardware
//
#include <hip/hip_runtime.h>
#include <cstdint>
#include <cstddef>

// ---------------- problem constants ----------------
#define B_   2
#define S_   1024
#define T_   2048          // B_*S_ tokens
#define DM   1024          // D_MODEL
#define DI   2048          // D_INNER
#define DST  16            // D_STATE
#define DE   2048          // D_EXPERT
#define NE   8             // N_EXPERTS
#define CH   32            // scan chunks
#define CL   32            // scan chunk length (CH*CL == S_)
#define THRESH_ 1.0f
#define LEAK_   0.9f

typedef __attribute__((ext_vector_type(16))) __bf16 v16bf;
typedef __attribute__((ext_vector_type(8)))  float  v8f;

// async global->LDS path (CDNA5) if the toolchain exposes it; else manual copy
#if defined(__gfx1250__) && __has_builtin(__builtin_amdgcn_global_load_async_to_lds_b128)
#define ASYNC_LDS 1
// match the builtin's parameter types exactly (probe-decoded):
//   param0: AS(1) int __vector_size__(16) *   (global src)
//   param1: AS(3) int __vector_size__(16) *   (LDS dst)
typedef int v4i_raw __attribute__((vector_size(16)));
typedef __attribute__((address_space(1))) v4i_raw as1_v4i;
typedef __attribute__((address_space(3))) v4i_raw as3_v4i;
#else
#define ASYNC_LDS 0
#endif

__device__ __forceinline__ void async_wait_all() {
#if ASYNC_LDS
#if __has_builtin(__builtin_amdgcn_s_wait_asynccnt)
  __builtin_amdgcn_s_wait_asynccnt(0);
#else
  asm volatile("s_wait_asynccnt 0" ::: "memory");
#endif
#endif
}

// ---------------- small helpers ----------------
__device__ __forceinline__ unsigned short f2bf(float f) {
  union { float f; unsigned u; } x; x.f = f;
  unsigned u = x.u;
  unsigned r = (u + 0x7FFFu + ((u >> 16) & 1u)) >> 16;   // RNE
  return (unsigned short)r;
}
__device__ __forceinline__ float bf2f(unsigned short h) {
  union { unsigned u; float f; } x; x.u = ((unsigned)h) << 16; return x.f;
}
// pack two f32 -> two bf16 (truncate) in ONE v_perm_b32: dst = {hi16(odd), hi16(even)}
__device__ __forceinline__ unsigned pkbf(float even, float odd) {
  return __builtin_amdgcn_perm(__float_as_uint(odd), __float_as_uint(even), 0x07060302u);
}
__device__ __forceinline__ float siluf(float x)     { return x / (1.f + __expf(-x)); }
__device__ __forceinline__ float softplusf(float x) { return x > 20.f ? x : log1pf(__expf(x)); }

union Frag { v16bf v; uint4 q[2]; };

// stage one 64x32 bf16 A tile row-chunk (16 elems) into As[row][k] (stride 40)
__device__ __forceinline__ void stage_a(const unsigned short* __restrict__ A, int K,
                                        int blockRow, int arow, int akh,
                                        unsigned short* __restrict__ As, int k0) {
  const unsigned short* s = A + (size_t)(blockRow + arow) * K + k0 + akh;
#if ASYNC_LDS
  __builtin_amdgcn_global_load_async_to_lds_b128(
      (as1_v4i*)s, (as3_v4i*)&As[arow * 40 + akh], 0, 0);
  __builtin_amdgcn_global_load_async_to_lds_b128(
      (as1_v4i*)(s + 8), (as3_v4i*)&As[arow * 40 + akh + 8], 0, 0);
#else
  uint4 p0 = *(const uint4*)s;
  uint4 p1 = *(const uint4*)(s + 8);
  *(uint4*)&As[arow * 40 + akh]     = p0;
  *(uint4*)&As[arow * 40 + akh + 8] = p1;
#endif
}

// stage a 32x64 f32 weight tile, transposed+converted, into Bs[col][k] (u16, stride 40)
__device__ __forceinline__ void stage_w(const float* __restrict__ W, int N, int k0,
                                        int blockCol, int bkp, int bng,
                                        unsigned short* __restrict__ Bs, int K) {
  const float* w0 = W + (size_t)(k0 + 2 * bkp) * N + blockCol + bng;
  const float* w1 = w0 + N;
  float4 e0 = *(const float4*)w0, e1 = *(const float4*)(w0 + 4);
  float4 o0 = *(const float4*)w1, o1 = *(const float4*)(w1 + 4);
  float ev[8] = { e0.x, e0.y, e0.z, e0.w, e1.x, e1.y, e1.z, e1.w };
  float od[8] = { o0.x, o0.y, o0.z, o0.w, o1.x, o1.y, o1.z, o1.w };
  #pragma unroll
  for (int j = 0; j < 8; j++)
    *(unsigned*)&Bs[(bng + j) * 40 + 2 * bkp] = pkbf(ev[j], od[j]);
  if (k0 + 64 < K) __builtin_prefetch((const void*)(w0 + (size_t)64 * N), 0, 0);
}

__device__ __forceinline__ void load_afrag(const unsigned short* __restrict__ As,
                                           int r0, int kb, Frag& a0, Frag& a1) {
  a0.q[0] = *(const uint4*)&As[r0 * 40 + kb];
  a0.q[1] = *(const uint4*)&As[r0 * 40 + kb + 8];
  a1.q[0] = *(const uint4*)&As[(r0 + 16) * 40 + kb];
  a1.q[1] = *(const uint4*)&As[(r0 + 16) * 40 + kb + 8];
}
__device__ __forceinline__ void load_bfrag(const unsigned short* __restrict__ Bs,
                                           int c0, int kb2, Frag& b0, Frag& b1) {
  b0.q[0] = *(const uint4*)&Bs[c0 * 40 + kb2];
  b0.q[1] = *(const uint4*)&Bs[c0 * 40 + kb2 + 8];
  b1.q[0] = *(const uint4*)&Bs[(c0 + 16) * 40 + kb2];
  b1.q[1] = *(const uint4*)&Bs[(c0 + 16) * 40 + kb2 + 8];
}

// ---------------- WMMA GEMM: C[M,N] = A(bf16)[M,K] @ W(f32)[K,N] ----------------
// 64x64 block tile, 4 waves, 2x2 WMMA 16x16x32 bf16 per wave, double-buffered LDS.
// MODE 0: C = acc     MODE 1: C = acc + R     MODE 3: MoE down scatter-add
template<int MODE>
__global__ __launch_bounds__(128)
void k_gemm(const unsigned short* __restrict__ A, const float* __restrict__ W,
            float* __restrict__ C, int M, int N, int K, int ldc,
            const float* __restrict__ R, const int* __restrict__ countp,
            const int* __restrict__ toks, const float* __restrict__ wfull, int expert)
{
  int cnt = M;
  if (countp) { cnt = *countp; if ((int)blockIdx.y * 64 >= cnt) return; }

  __shared__ __align__(16) unsigned short As[2][64 * 40];
  __shared__ __align__(16) unsigned short Bs[2][64 * 40];

  const int tid  = threadIdx.x;
  const int lane = tid & 31;
  const int wave = tid >> 5;
  const int wm = wave >> 1, wn = wave & 1;
  const int blockRow = blockIdx.y * 64;
  const int blockCol = blockIdx.x * 64;

  v8f acc00 = {}, acc01 = {}, acc10 = {}, acc11 = {};

  const int arow = tid >> 1, akh = (tid & 1) * 16;
  const int bkp  = tid >> 3, bng = (tid & 7) * 8;
  const int r0   = wm * 32 + (lane & 15);
  const int kb   = (lane < 16) ? 0 : 16;
  const int c0   = wn * 32 + (lane & 15);
  const int kb2  = (lane < 16) ? 0 : 16;

  stage_a(A, K, blockRow, arow, akh, As[0], 0);
  stage_w(W, N, 0, blockCol, bkp, bng, Bs[0], K);
  async_wait_all();
  __syncthreads();

  for (int k0 = 0; k0 < K; k0 += 32) {
    const int cur = (k0 >> 5) & 1, nxt = cur ^ 1;
    if (k0 + 32 < K) {
      stage_a(A, K, blockRow, arow, akh, As[nxt], k0 + 32);
      stage_w(W, N, k0 + 32, blockCol, bkp, bng, Bs[nxt], K);
    }
    Frag a0, a1, b0, b1;
    load_afrag(As[cur], r0, kb, a0, a1);
    load_bfrag(Bs[cur], c0, kb2, b0, b1);
    acc00 = __builtin_amdgcn_wmma_f32_16x16x32_bf16(false, a0.v, false, b0.v, (short)0, acc00, false, false);
    acc01 = __builtin_amdgcn_wmma_f32_16x16x32_bf16(false, a0.v, false, b1.v, (short)0, acc01, false, false);
    acc10 = __builtin_amdgcn_wmma_f32_16x16x32_bf16(false, a1.v, false, b0.v, (short)0, acc10, false, false);
    acc11 = __builtin_amdgcn_wmma_f32_16x16x32_bf16(false, a1.v, false, b1.v, (short)0, acc11, false, false);
    async_wait_all();
    __syncthreads();
  }

  const int rowsel = (lane < 16) ? 0 : 8;
  const int colw   = lane & 15;
  v8f accs[2][2] = { { acc00, acc01 }, { acc10, acc11 } };
  #pragma unroll
  for (int mi = 0; mi < 2; mi++)
    #pragma unroll
    for (int ni = 0; ni < 2; ni++)
      #pragma unroll
      for (int v = 0; v < 8; v++) {
        int row = blockRow + wm * 32 + mi * 16 + rowsel + v;
        int col = blockCol + wn * 32 + ni * 16 + colw;
        float val = accs[mi][ni][v];
        if (MODE == 0) {
          C[(size_t)row * ldc + col] = val;
        } else if (MODE == 1) {
          C[(size_t)row * ldc + col] = val + R[(size_t)row * ldc + col];
        } else {
          if (row < cnt) {
            int t = toks[row];
            C[(size_t)t * ldc + col] += wfull[(size_t)t * NE + expert] * val;
          }
        }
      }
}

// ---------------- fused MoE gate+up: H = bf16(silu(A@Wg) * (A@Wu)) ----------------
__global__ __launch_bounds__(128)
void k_moe_gu(const unsigned short* __restrict__ A, const float* __restrict__ Wg,
              const float* __restrict__ Wu, unsigned short* __restrict__ H,
              int N, int K, const int* __restrict__ countp)
{
  int cnt = *countp;
  if ((int)blockIdx.y * 64 >= cnt) return;

  __shared__ __align__(16) unsigned short As[2][64 * 40];
  __shared__ __align__(16) unsigned short Bg[2][64 * 40];
  __shared__ __align__(16) unsigned short Bu[2][64 * 40];

  const int tid  = threadIdx.x;
  const int lane = tid & 31;
  const int wave = tid >> 5;
  const int wm = wave >> 1, wn = wave & 1;
  const int blockRow = blockIdx.y * 64;
  const int blockCol = blockIdx.x * 64;

  v8f ag00 = {}, ag01 = {}, ag10 = {}, ag11 = {};
  v8f au00 = {}, au01 = {}, au10 = {}, au11 = {};

  const int arow = tid >> 1, akh = (tid & 1) * 16;
  const int bkp  = tid >> 3, bng = (tid & 7) * 8;
  const int r0   = wm * 32 + (lane & 15);
  const int kb   = (lane < 16) ? 0 : 16;
  const int c0   = wn * 32 + (lane & 15);
  const int kb2  = (lane < 16) ? 0 : 16;

  stage_a(A, K, blockRow, arow, akh, As[0], 0);
  stage_w(Wg, N, 0, blockCol, bkp, bng, Bg[0], K);
  stage_w(Wu, N, 0, blockCol, bkp, bng, Bu[0], K);
  async_wait_all();
  __syncthreads();

  for (int k0 = 0; k0 < K; k0 += 32) {
    const int cur = (k0 >> 5) & 1, nxt = cur ^ 1;
    if (k0 + 32 < K) {
      stage_a(A, K, blockRow, arow, akh, As[nxt], k0 + 32);
      stage_w(Wg, N, k0 + 32, blockCol, bkp, bng, Bg[nxt], K);
      stage_w(Wu, N, k0 + 32, blockCol, bkp, bng, Bu[nxt], K);
    }
    Frag a0, a1, g0, g1, u0, u1;
    load_afrag(As[cur], r0, kb, a0, a1);
    load_bfrag(Bg[cur], c0, kb2, g0, g1);
    load_bfrag(Bu[cur], c0, kb2, u0, u1);
    ag00 = __builtin_amdgcn_wmma_f32_16x16x32_bf16(false, a0.v, false, g0.v, (short)0, ag00, false, false);
    ag01 = __builtin_amdgcn_wmma_f32_16x16x32_bf16(false, a0.v, false, g1.v, (short)0, ag01, false, false);
    ag10 = __builtin_amdgcn_wmma_f32_16x16x32_bf16(false, a1.v, false, g0.v, (short)0, ag10, false, false);
    ag11 = __builtin_amdgcn_wmma_f32_16x16x32_bf16(false, a1.v, false, g1.v, (short)0, ag11, false, false);
    au00 = __builtin_amdgcn_wmma_f32_16x16x32_bf16(false, a0.v, false, u0.v, (short)0, au00, false, false);
    au01 = __builtin_amdgcn_wmma_f32_16x16x32_bf16(false, a0.v, false, u1.v, (short)0, au01, false, false);
    au10 = __builtin_amdgcn_wmma_f32_16x16x32_bf16(false, a1.v, false, u0.v, (short)0, au10, false, false);
    au11 = __builtin_amdgcn_wmma_f32_16x16x32_bf16(false, a1.v, false, u1.v, (short)0, au11, false, false);
    async_wait_all();
    __syncthreads();
  }

  const int rowsel = (lane < 16) ? 0 : 8;
  const int colw   = lane & 15;
  v8f gs[2][2] = { { ag00, ag01 }, { ag10, ag11 } };
  v8f us[2][2] = { { au00, au01 }, { au10, au11 } };
  #pragma unroll
  for (int mi = 0; mi < 2; mi++)
    #pragma unroll
    for (int ni = 0; ni < 2; ni++)
      #pragma unroll
      for (int v = 0; v < 8; v++) {
        int row = blockRow + wm * 32 + mi * 16 + rowsel + v;
        int col = blockCol + wn * 32 + ni * 16 + colw;
        H[(size_t)row * N + col] = f2bf(siluf(gs[mi][ni][v]) * us[mi][ni][v]);
      }
}

// ---------------- rmsnorm -> bf16 ----------------
__global__ void k_rmsnorm(const float* __restrict__ x, const float* __restrict__ w,
                          unsigned short* __restrict__ out) {
  int row = blockIdx.x;
  const float* xr = x + (size_t)row * DM;
  float ss = 0.f;
  for (int i = threadIdx.x; i < DM; i += 256) { float v = xr[i]; ss += v * v; }
  __shared__ float red[256];
  red[threadIdx.x] = ss; __syncthreads();
  for (int off = 128; off > 0; off >>= 1) {
    if (threadIdx.x < off) red[threadIdx.x] += red[threadIdx.x + off];
    __syncthreads();
  }
  float scale = rsqrtf(red[0] / (float)DM + 1e-6f);
  for (int i = threadIdx.x; i < DM; i += 256)
    out[(size_t)row * DM + i] = f2bf(xr[i] * scale * w[i]);
}

// ---------------- causal depthwise conv (k=4) + silu ----------------
__global__ void k_conv(const float* __restrict__ xz, const float* __restrict__ cw,
                       const float* __restrict__ cb, float* __restrict__ xc) {
  int idx = blockIdx.x * 256 + threadIdx.x;
  if (idx >= T_ * DI) return;
  int d = idx % DI, t = idx / DI, s = t % S_, b = t / S_;
  float acc = cb[d];
  #pragma unroll
  for (int k = 0; k < 4; k++) {
    int sp = s - 3 + k;
    if (sp >= 0) acc += xz[((size_t)(b * S_ + sp)) * (2 * DI) + d] * cw[d * 4 + k];
  }
  xc[idx] = siluf(acc);
}

// ---------------- x_proj: xdbl[T,33] = xc @ x_proj_w ----------------
__global__ void k_xproj(const float* __restrict__ xc, const float* __restrict__ pw,
                        float* __restrict__ xdbl) {
  int t = blockIdx.x;
  int wv = threadIdx.x >> 5, l = threadIdx.x & 31;
  const float* xr = xc + (size_t)t * DI;
  for (int col = wv; col < 33; col += 8) {
    float s = 0.f;
    for (int k = l; k < DI; k += 32) s += xr[k] * pw[(size_t)k * 33 + col];
    for (int off = 16; off > 0; off >>= 1) s += __shfl_down(s, off, 32);
    if (l == 0) xdbl[(size_t)t * 33 + col] = s;
  }
}

// ---------------- chunked selective scan (affine scan over S) ----------------
__global__ void k_scan1(const float* __restrict__ xdbl, const float* __restrict__ xc,
                        const float* __restrict__ A_log, const float* __restrict__ dt_w,
                        const float* __restrict__ dt_b,
                        float* __restrict__ ASUM, float* __restrict__ BSUM) {
  int gid = blockIdx.x * 256 + threadIdx.x;
  if (gid >= B_ * CH * DI) return;
  int d = gid % DI, bc = gid / DI, c = bc % CH, b = bc / CH;
  float An[DST];
  #pragma unroll
  for (int n = 0; n < DST; n++) An[n] = -__expf(A_log[(size_t)d * DST + n]);
  float dtw = dt_w[d], dtb = dt_b[d];
  float Ap[DST], Bp[DST];
  #pragma unroll
  for (int n = 0; n < DST; n++) { Ap[n] = 1.f; Bp[n] = 0.f; }
  for (int s = c * CL; s < (c + 1) * CL; s++) {
    size_t t = (size_t)b * S_ + s;
    const float* xd = xdbl + t * 33;
    float dt  = softplusf(xd[0] * dtw + dtb);
    float dtx = dt * xc[t * DI + d];
    #pragma unroll
    for (int n = 0; n < DST; n++) {
      float a = __expf(dt * An[n]);
      Bp[n] = a * Bp[n] + dtx * xd[1 + n];
      Ap[n] *= a;
    }
  }
  size_t base = (((size_t)b * DI + d) * CH + c) * DST;
  #pragma unroll
  for (int n = 0; n < DST; n++) { ASUM[base + n] = Ap[n]; BSUM[base + n] = Bp[n]; }
}

__global__ void k_scan2(const float* __restrict__ ASUM, const float* __restrict__ BSUM,
                        float* __restrict__ HINIT) {
  int gid = blockIdx.x * 256 + threadIdx.x;
  if (gid >= B_ * DI) return;
  size_t base = (size_t)gid * CH * DST;
  float h[DST];
  #pragma unroll
  for (int n = 0; n < DST; n++) h[n] = 0.f;
  for (int c = 0; c < CH; c++) {
    size_t idx = base + (size_t)c * DST;
    #pragma unroll
    for (int n = 0; n < DST; n++) {
      HINIT[idx + n] = h[n];
      h[n] = ASUM[idx + n] * h[n] + BSUM[idx + n];
    }
  }
}

__global__ void k_scan3(const float* __restrict__ xdbl, const float* __restrict__ xc,
                        const float* __restrict__ A_log, const float* __restrict__ dt_w,
                        const float* __restrict__ dt_b, const float* __restrict__ HINIT,
                        float* __restrict__ y) {
  int gid = blockIdx.x * 256 + threadIdx.x;
  if (gid >= B_ * CH * DI) return;
  int d = gid % DI, bc = gid / DI, c = bc % CH, b = bc / CH;
  float An[DST];
  #pragma unroll
  for (int n = 0; n < DST; n++) An[n] = -__expf(A_log[(size_t)d * DST + n]);
  float dtw = dt_w[d], dtb = dt_b[d];
  float h[DST];
  size_t hbase = (((size_t)b * DI + d) * CH + c) * DST;
  #pragma unroll
  for (int n = 0; n < DST; n++) h[n] = HINIT[hbase + n];
  for (int s = c * CL; s < (c + 1) * CL; s++) {
    size_t t = (size_t)b * S_ + s;
    const float* xd = xdbl + t * 33;
    float dt  = softplusf(xd[0] * dtw + dtb);
    float dtx = dt * xc[t * DI + d];
    float ya = 0.f;
    #pragma unroll
    for (int n = 0; n < DST; n++) {
      float a = __expf(dt * An[n]);
      h[n] = a * h[n] + dtx * xd[1 + n];
      ya  += h[n] * xd[17 + n];
    }
    y[t * DI + d] = ya;
  }
}

// ---------------- SSM output gating -> bf16 ----------------
__global__ void k_gatemul(const float* __restrict__ y, const float* __restrict__ xc,
                          const float* __restrict__ xz, const float* __restrict__ Dskip,
                          unsigned short* __restrict__ yg) {
  int idx = blockIdx.x * 256 + threadIdx.x;
  if (idx >= T_ * DI) return;
  int d = idx % DI, t = idx / DI;
  float zv = xz[(size_t)t * (2 * DI) + DI + d];
  yg[idx] = f2bf((y[idx] + xc[idx] * Dskip[d]) * siluf(zv));
}

// ---------------- LIF scan: one thread per (b,d) ----------------
__global__ void k_lif(const float* __restrict__ cur, unsigned short* __restrict__ spikes,
                      float* __restrict__ membrane) {
  int gid = blockIdx.x * 256 + threadIdx.x;
  if (gid >= B_ * DM) return;
  int b = gid / DM, d = gid % DM;
  float m = 0.f;
  for (int s = 0; s < S_; s++) {
    size_t t = (size_t)b * S_ + s;
    m = LEAK_ * m + cur[t * DM + d];
    float sp = (m >= THRESH_) ? 1.f : 0.f;
    spikes[t * DM + d] = f2bf(sp);
    m -= sp * THRESH_;
  }
  membrane[(size_t)b * DM + d] = m;
}

// ---------------- MoE routing ----------------
__global__ void k_zero_counts(int* c) { if (threadIdx.x < NE) c[threadIdx.x] = 0; }

__global__ void k_router(const unsigned short* __restrict__ xn, const float* __restrict__ rw,
                         float* __restrict__ wfull, int* __restrict__ counts,
                         int* __restrict__ toks) {
  int t = blockIdx.x * 256 + threadIdx.x;
  if (t >= T_) return;
  float lg[NE] = {0, 0, 0, 0, 0, 0, 0, 0};
  const unsigned short* xr = xn + (size_t)t * DM;
  for (int k = 0; k < DM; k++) {
    float xv = bf2f(xr[k]);
    const float* r = rw + (size_t)k * NE;
    #pragma unroll
    for (int e = 0; e < NE; e++) lg[e] += xv * r[e];
  }
  float mx = lg[0];
  for (int e = 1; e < NE; e++) mx = fmaxf(mx, lg[e]);
  float p[NE], sum = 0.f;
  for (int e = 0; e < NE; e++) { p[e] = __expf(lg[e] - mx); sum += p[e]; }
  for (int e = 0; e < NE; e++) p[e] /= sum;
  int i0 = 0; float p0 = p[0];
  for (int e = 1; e < NE; e++) if (p[e] > p0) { p0 = p[e]; i0 = e; }
  int i1 = -1; float p1 = -1.f;
  for (int e = 0; e < NE; e++) if (e != i0 && p[e] > p1) { p1 = p[e]; i1 = e; }
  for (int e = 0; e < NE; e++) wfull[(size_t)t * NE + e] = (e == i0 || e == i1) ? p[e] : 0.f;
  int pos0 = atomicAdd(&counts[i0], 1); toks[i0 * T_ + pos0] = t;
  int pos1 = atomicAdd(&counts[i1], 1); toks[i1 * T_ + pos1] = t;
}

__global__ void k_gather(const unsigned short* __restrict__ xn, const int* __restrict__ countp,
                         const int* __restrict__ toks, unsigned short* __restrict__ Ag) {
  int idx = blockIdx.x * 256 + threadIdx.x;
  if (idx >= T_ * DM) return;
  int r = idx / DM, c = idx % DM;
  int cnt = *countp;
  Ag[idx] = (r < cnt) ? xn[(size_t)toks[r] * DM + c] : (unsigned short)0;
}

// ---------------- workspace layout (bytes) ----------------
#define MB (1ull << 20)
static const size_t OFF_XN    = 0;            // bf16 T*DM    4MB
static const size_t OFF_XZ    = 4   * MB;     // f32  T*2DI  32MB
static const size_t OFF_XC    = 36  * MB;     // f32  T*DI   16MB
static const size_t OFF_XDBL  = 52  * MB;     // f32  T*33   <1MB
static const size_t OFF_Y     = 53  * MB;     // f32  T*DI   16MB
static const size_t OFF_YG    = 69  * MB;     // bf16 T*DI    8MB
static const size_t OFF_X1    = 77  * MB;     // f32  T*DM    8MB
static const size_t OFF_XN2   = 85  * MB;     // bf16 T*DM    4MB
static const size_t OFF_CUR   = 89  * MB;     // f32  T*DM    8MB
static const size_t OFF_SPK   = 97  * MB;     // bf16 T*DM    4MB
static const size_t OFF_XN3   = 101 * MB;     // bf16 T*DM    4MB
static const size_t OFF_WF    = 105 * MB;     // f32  T*NE   64KB
static const size_t OFF_CNT   = 105 * MB + 128 * 1024;
static const size_t OFF_TOK   = 105 * MB + 256 * 1024;
static const size_t OFF_AG    = 106 * MB;     // bf16 T*DM    4MB
static const size_t OFF_H     = 110 * MB;     // bf16 T*DE    8MB
static const size_t OFF_ASUM  = 118 * MB;     // f32 B*DI*CH*DST  8MB
static const size_t OFF_BSUM  = 134 * MB;     // f32              8MB
static const size_t OFF_HINIT = 150 * MB;     // f32              8MB

extern "C" void kernel_launch(void* const* d_in, const int* in_sizes, int n_in,
                              void* d_out, int out_size, void* d_ws, size_t ws_size,
                              hipStream_t stream) {
  const float* x       = (const float*)d_in[0];
  const float* ssm_nw  = (const float*)d_in[1];
  const float* in_w    = (const float*)d_in[2];
  const float* conv_w  = (const float*)d_in[3];
  const float* conv_b  = (const float*)d_in[4];
  const float* A_log   = (const float*)d_in[5];
  const float* Dskip   = (const float*)d_in[6];
  const float* xproj_w = (const float*)d_in[7];
  const float* dt_w    = (const float*)d_in[8];
  const float* dt_b    = (const float*)d_in[9];
  const float* out_w   = (const float*)d_in[10];
  const float* lif_nw  = (const float*)d_in[11];
  const float* lif_iw  = (const float*)d_in[12];
  const float* lif_ow  = (const float*)d_in[13];
  const float* moe_nw  = (const float*)d_in[14];
  const float* rout_w  = (const float*)d_in[15];
  const float* gate_w  = (const float*)d_in[16];
  const float* up_w    = (const float*)d_in[17];
  const float* down_w  = (const float*)d_in[18];

  float* xout = (float*)d_out;                       // [T, DM]
  float* memb = xout + (size_t)T_ * DM;              // [B, DM]

  char* ws = (char*)d_ws;
  unsigned short* XN   = (unsigned short*)(ws + OFF_XN);
  float*          XZ   = (float*)(ws + OFF_XZ);
  float*          XC   = (float*)(ws + OFF_XC);
  float*          XDB  = (float*)(ws + OFF_XDBL);
  float*          Y    = (float*)(ws + OFF_Y);
  unsigned short* YG   = (unsigned short*)(ws + OFF_YG);
  float*          X1   = (float*)(ws + OFF_X1);
  unsigned short* XN2  = (unsigned short*)(ws + OFF_XN2);
  float*          CUR  = (float*)(ws + OFF_CUR);
  unsigned short* SPK  = (unsigned short*)(ws + OFF_SPK);
  unsigned short* XN3  = (unsigned short*)(ws + OFF_XN3);
  float*          WF   = (float*)(ws + OFF_WF);
  int*            CNT  = (int*)(ws + OFF_CNT);
  int*            TOK  = (int*)(ws + OFF_TOK);
  unsigned short* AG   = (unsigned short*)(ws + OFF_AG);
  unsigned short* H    = (unsigned short*)(ws + OFF_H);
  float*          ASUM = (float*)(ws + OFF_ASUM);
  float*          BSUM = (float*)(ws + OFF_BSUM);
  float*          HIN  = (float*)(ws + OFF_HINIT);

  dim3 gblk(128);

  // ===== SSM =====
  k_rmsnorm<<<T_, 256, 0, stream>>>(x, ssm_nw, XN);
  k_gemm<0><<<dim3((2 * DI) / 64, T_ / 64), gblk, 0, stream>>>(
      XN, in_w, XZ, T_, 2 * DI, DM, 2 * DI, nullptr, nullptr, nullptr, nullptr, 0);
  k_conv<<<(T_ * DI) / 256, 256, 0, stream>>>(XZ, conv_w, conv_b, XC);
  k_xproj<<<T_, 256, 0, stream>>>(XC, xproj_w, XDB);
  k_scan1<<<(B_ * CH * DI) / 256, 256, 0, stream>>>(XDB, XC, A_log, dt_w, dt_b, ASUM, BSUM);
  k_scan2<<<(B_ * DI + 255) / 256, 256, 0, stream>>>(ASUM, BSUM, HIN);
  k_scan3<<<(B_ * CH * DI) / 256, 256, 0, stream>>>(XDB, XC, A_log, dt_w, dt_b, HIN, Y);
  k_gatemul<<<(T_ * DI) / 256, 256, 0, stream>>>(Y, XC, XZ, Dskip, YG);
  k_gemm<1><<<dim3(DM / 64, T_ / 64), gblk, 0, stream>>>(
      YG, out_w, X1, T_, DM, DI, DM, x, nullptr, nullptr, nullptr, 0);

  // ===== LIF =====
  k_rmsnorm<<<T_, 256, 0, stream>>>(X1, lif_nw, XN2);
  k_gemm<0><<<dim3(DM / 64, T_ / 64), gblk, 0, stream>>>(
      XN2, lif_iw, CUR, T_, DM, DM, DM, nullptr, nullptr, nullptr, nullptr, 0);
  k_lif<<<(B_ * DM + 255) / 256, 256, 0, stream>>>(CUR, SPK, memb);
  k_gemm<1><<<dim3(DM / 64, T_ / 64), gblk, 0, stream>>>(
      SPK, lif_ow, xout, T_, DM, DM, DM, X1, nullptr, nullptr, nullptr, 0);

  // ===== MoE (top-2 sparse dispatch) =====
  k_rmsnorm<<<T_, 256, 0, stream>>>(xout, moe_nw, XN3);
  k_zero_counts<<<1, 64, 0, stream>>>(CNT);
  k_router<<<(T_ + 255) / 256, 256, 0, stream>>>(XN3, rout_w, WF, CNT, TOK);

  for (int e = 0; e < NE; e++) {
    const int* cp = CNT + e;
    const int* tp = TOK + e * T_;
    k_gather<<<(T_ * DM) / 256, 256, 0, stream>>>(XN3, cp, tp, AG);
    k_moe_gu<<<dim3(DE / 64, T_ / 64), gblk, 0, stream>>>(
        AG, gate_w + (size_t)e * DM * DE, up_w + (size_t)e * DM * DE, H, DE, DM, cp);
    k_gemm<3><<<dim3(DM / 64, T_ / 64), gblk, 0, stream>>>(
        H, down_w + (size_t)e * DE * DM, xout, T_, DM, DE, DM, nullptr, cp, tp, WF, e);
  }
}
